// VectorQuantizer_1589137899912
// MI455X (gfx1250) — matrix operand order repaired and checked by reading, not compiled, not run on hardware
//
#include <hip/hip_runtime.h>

typedef __attribute__((ext_vector_type(16))) _Float16 v16h;
typedef __attribute__((ext_vector_type(8)))  float    v8f;

#define D_DIM   64
#define K_DIM   1024
#define TILES   (K_DIM / 16)          // 64 code tiles
#define BP_HALF (TILES * 2 * 32 * 16) // 65536 halves = 128KB B-pack

// ---------------------------------------------------------------------------
// Prep 1: pack codebook [64,1024] f32 -> f16 WMMA B-operand layout.
// B tile is 32x16 (KxN) f16. Per lane: n = lane&15, hi = lane>>4,
// halves i=0..15 hold B[hi*16 + i][n], i.e. emb[d = s*32 + hi*16 + i][code].
// Flat layout: bpack[((t*2+s)*32 + lane)*16 + i]
// ---------------------------------------------------------------------------
__global__ void vq_prep_bpack(const float* __restrict__ emb,
                              _Float16* __restrict__ bpack) {
    int idx = blockIdx.x * 256 + threadIdx.x;      // 0..65535
    int i   = idx & 15;
    int l   = (idx >> 4) & 31;
    int ts  = idx >> 9;                            // t*2+s, 0..127
    int t   = ts >> 1;
    int s   = ts & 1;
    int n   = l & 15;
    int hi  = l >> 4;
    int d   = s * 32 + hi * 16 + i;
    int code = t * 16 + n;
    bpack[idx] = (_Float16)emb[d * K_DIM + code];
}

// Prep 2: per-code squared norms ||e_k||^2 (f32 exact)
__global__ void vq_prep_norms(const float* __restrict__ emb,
                              float* __restrict__ norms) {
    int k = blockIdx.x * 256 + threadIdx.x;
    if (k >= K_DIM) return;
    float s = 0.f;
    #pragma unroll
    for (int d = 0; d < D_DIM; ++d) {
        float v = emb[d * K_DIM + k];
        s = fmaf(v, v, s);
    }
    norms[k] = s;
}

// CDNA5 async bulk copy: memory -> LDS without touching VGPRs (ASYNCcnt).
__device__ __forceinline__ void async_b128_to_lds(unsigned ldsOff, const void* gaddr) {
    asm volatile("global_load_async_to_lds_b128 %0, %1, off"
                 :: "v"(ldsOff), "v"(gaddr) : "memory");
}
__device__ __forceinline__ void wait_asynccnt0() {
    asm volatile("s_wait_asynccnt 0" ::: "memory");
}

// ---------------------------------------------------------------------------
// Main: 8 waves/block, each wave owns 64 rows (4 row-tiles of 16).
// Distances via v_wmma_f32_16x16x32_f16. Argmin: code index packed into the
// low 10 mantissa bits of the score (v_and_or_b32) so a single min tracks
// value+index; two tiles processed per iteration so the running min fuses to
// v_min3_num_f32. Quantized output gathered in exact fp32.
// ---------------------------------------------------------------------------
__global__ void __launch_bounds__(256)
vq_main(const float* __restrict__ x,
        const float* __restrict__ emb,
        const _Float16* __restrict__ bpack,
        const float* __restrict__ norms,
        float* __restrict__ out,
        float* __restrict__ lossOut,
        float lossScale) {
    __shared__ __align__(16) _Float16 ldsB[BP_HALF];   // 128 KB
    __shared__ __align__(16) float    ldsN[K_DIM];     //   4 KB

    const int tid = threadIdx.x;

    // Async stage of B-pack + norms into LDS (L2 -> LDS, ASYNCcnt-tracked).
    {
        const unsigned ldsBBase = (unsigned)(uintptr_t)&ldsB[0];
        const unsigned ldsNBase = (unsigned)(uintptr_t)&ldsN[0];
        const char* gB = (const char*)bpack;
        const char* gN = (const char*)norms;
        #pragma unroll
        for (int i = 0; i < (BP_HALF * 2) / (16 * 256); ++i) {   // 32 chunks/thread
            int off = (i * 256 + tid) * 16;
            async_b128_to_lds(ldsBBase + off, gB + off);
        }
        async_b128_to_lds(ldsNBase + tid * 16, gN + tid * 16);   // 4KB norms
        wait_asynccnt0();
    }
    __syncthreads();

    const int lane   = tid & 31;
    const int wave   = tid >> 5;
    const int m      = lane & 15;   // row within tile (A) / code within tile (B)
    const int hi     = lane >> 4;   // half-wave select
    const size_t rowBase = ((size_t)blockIdx.x * 8 + wave) * 64;

    // Build A operands: 4 row-tiles x 2 K-steps, f16 in registers.
    v16h A[4][2];
    #pragma unroll
    for (int r = 0; r < 4; ++r) {
        const size_t row = rowBase + (size_t)r * 16 + m;
        #pragma unroll
        for (int s = 0; s < 2; ++s) {
            const float* xr = x + row * D_DIM + s * 32;
            float tmp[16];
            *(float4*)&tmp[0]  = *(const float4*)(xr + hi * 16);
            *(float4*)&tmp[4]  = *(const float4*)(xr + hi * 16 + 4);
            *(float4*)&tmp[8]  = *(const float4*)(xr + hi * 16 + 8);
            *(float4*)&tmp[12] = *(const float4*)(xr + hi * 16 + 12);
            #pragma unroll
            for (int i = 0; i < 16; ++i) A[r][s][i] = (_Float16)tmp[i];
        }
    }

    // Running packed (score|index) minima: 8 rows per row-tile per lane.
    float best[4][8];
    #pragma unroll
    for (int r = 0; r < 4; ++r)
        #pragma unroll
        for (int j = 0; j < 8; ++j) best[r][j] = __builtin_inff();

    // Sweep 64 code tiles two at a time: B from LDS, 16 WMMAs per iteration.
    for (int t = 0; t < TILES; t += 2) {
        const v16h b0a = *reinterpret_cast<const v16h*>(&ldsB[(t * 2 + 0) * 512 + lane * 16]);
        const v16h b1a = *reinterpret_cast<const v16h*>(&ldsB[(t * 2 + 1) * 512 + lane * 16]);
        const v16h b0b = *reinterpret_cast<const v16h*>(&ldsB[(t * 2 + 2) * 512 + lane * 16]);
        const v16h b1b = *reinterpret_cast<const v16h*>(&ldsB[(t * 2 + 3) * 512 + lane * 16]);
        const unsigned kA = (unsigned)(t * 16 + m);
        const unsigned kB = kA + 16u;
        const float nrmA = ldsN[t * 16 + m];
        const float nrmB = ldsN[t * 16 + 16 + m];

        #pragma unroll
        for (int r = 0; r < 4; ++r) {
            v8f ca = {0.f, 0.f, 0.f, 0.f, 0.f, 0.f, 0.f, 0.f};
            v8f cb = {0.f, 0.f, 0.f, 0.f, 0.f, 0.f, 0.f, 0.f};
            ca = __builtin_amdgcn_wmma_f32_16x16x32_f16(false, A[r][0], false, b0a,
                                                        (short)0, ca, false, false);
            cb = __builtin_amdgcn_wmma_f32_16x16x32_f16(false, A[r][0], false, b0b,
                                                        (short)0, cb, false, false);
            ca = __builtin_amdgcn_wmma_f32_16x16x32_f16(false, A[r][1], false, b1a,
                                                        (short)0, ca, false, false);
            cb = __builtin_amdgcn_wmma_f32_16x16x32_f16(false, A[r][1], false, b1b,
                                                        (short)0, cb, false, false);
            // score_k = ||e_k||^2 - 2*dot ; pack k into low 10 mantissa bits;
            // fold both tiles into the running min via v_min3_num_f32.
            #pragma unroll
            for (int j = 0; j < 8; ++j) {
                float sa = fmaf(-2.0f, ca[j], nrmA);
                float sb = fmaf(-2.0f, cb[j], nrmB);
                unsigned pa = (__float_as_uint(sa) & 0xFFFFFC00u) | kA;  // v_and_or_b32
                unsigned pb = (__float_as_uint(sb) & 0xFFFFFC00u) | kB;
                best[r][j] = fminf(best[r][j],
                                   fminf(__uint_as_float(pa), __uint_as_float(pb)));
            }
        }
    }

    // Epilogue: min-reduce packed keys across the 16-lane half, extract the
    // winning index, gather exact f32 code, write output, accumulate sq-error.
    float lsum = 0.f;
    #pragma unroll
    for (int r = 0; r < 4; ++r) {
        #pragma unroll
        for (int j = 0; j < 8; ++j) {
            float b = best[r][j];
            #pragma unroll
            for (int off = 1; off < 16; off <<= 1)
                b = fminf(b, __shfl_xor(b, off, 32));
            const unsigned k = __float_as_uint(b) & 0x3FFu;   // uniform per half
            // lanes 0-15 -> row j, lanes 16-31 -> row j+8
            const size_t row = rowBase + (size_t)r * 16 + hi * 8 + j;
            #pragma unroll
            for (int u = 0; u < 4; ++u) {
                const int dd = m + u * 16;
                const size_t off64 = row * D_DIM + dd;
                float e  = emb[(size_t)dd * K_DIM + k];
                float xv = x[off64];
                out[off64] = e;
                float df = e - xv;
                lsum = fmaf(df, df, lsum);
            }
        }
    }

    // Wave-reduce loss, single atomic per wave.
    #pragma unroll
    for (int off = 16; off >= 1; off >>= 1) lsum += __shfl_xor(lsum, off, 32);
    if (lane == 0) atomicAdd(lossOut, lsum * lossScale);
}

// ---------------------------------------------------------------------------
extern "C" void kernel_launch(void* const* d_in, const int* in_sizes, int n_in,
                              void* d_out, int out_size, void* d_ws, size_t ws_size,
                              hipStream_t stream) {
    const float* x   = (const float*)d_in[0];
    const float* emb = (const float*)d_in[1];
    float* out     = (float*)d_out;
    float* lossPtr = out + (out_size - 1);

    _Float16* bpack = (_Float16*)d_ws;
    float*    norms = (float*)((char*)d_ws + BP_HALF * sizeof(_Float16));

    hipMemsetAsync(lossPtr, 0, sizeof(float), stream);

    vq_prep_bpack<<<BP_HALF / 256, 256, 0, stream>>>(emb, bpack);
    vq_prep_norms<<<K_DIM / 256, 256, 0, stream>>>(emb, norms);

    const int N = in_sizes[0] / D_DIM;        // 262144 rows
    const int blocks = N / 512;               // 64 rows/wave * 8 waves/block
    const float lossScale = 1.25f / (float)(out_size - 1);

    vq_main<<<blocks, 256, 0, stream>>>(x, emb, bpack, norms, out, lossPtr, lossScale);
}
